// CausalSelfAttention_6777458393730
// MI455X (gfx1250) — compile-verified, matrix operand order checked by reading, not run on hardware
//
#include <hip/hip_runtime.h>

#define DEVINL __device__ __forceinline__

typedef __attribute__((ext_vector_type(16))) __bf16 bf16x16;
typedef __attribute__((ext_vector_type(8)))  float  v8f;
typedef __attribute__((ext_vector_type(4)))  unsigned int u32x4;

constexpr int BB  = 4, SS = 2048, DD = 1024, HH = 8, KVHn = 2, HDn = 128;
constexpr int BSr = BB * SS;                 // 8192 token rows
constexpr int NQKV = DD + 2 * KVHn * HDn;    // 1536 fused QKV columns

union Frag { bf16x16 f; u32x4 q[2]; };

// Per CDNA5 ISA 16-bit A/B layout: lanes 0-15 take K[0..7]+K[16..23],
// lanes 16-31 take K[8..15]+K[24..31] of a contiguous 32-wide K row.
DEVINL bf16x16 load_frag(const __bf16* row, int hi) {
  Frag fr;
  fr.q[0] = *(const u32x4*)(row + (hi << 3));
  fr.q[1] = *(const u32x4*)(row + 16 + (hi << 3));
  return fr.f;
}

DEVINL __bf16 f2bf(float f) {
  unsigned u = __builtin_bit_cast(unsigned, f);
  u += 0x7fffu + ((u >> 16) & 1u);             // round-to-nearest-even
  unsigned short hv = (unsigned short)(u >> 16);
  return __builtin_bit_cast(__bf16, hv);
}

DEVINL v8f vzero() { v8f z = {0.f,0.f,0.f,0.f,0.f,0.f,0.f,0.f}; return z; }

DEVINL v8f wmma_bf16(bf16x16 a, bf16x16 b, v8f c) {
  return __builtin_amdgcn_wmma_f32_16x16x32_bf16(false, a, false, b, (short)0, c, false, false);
}

// ---- CDNA5 async global->LDS DMA (ASYNCcnt-tracked, bypasses VGPRs) ----
DEVINL unsigned lds_off(const void* p) {     // wave-relative LDS byte address
  return (unsigned)(unsigned long long)p;
}
// copy 32 contiguous bytes: global[sbase + voff .. +31] -> LDS[ldso .. +31]
DEVINL void async_cp32(unsigned ldso, unsigned voff, const void* sbase) {
  asm volatile(
      "global_load_async_to_lds_b128 %0, %1, %2\n\t"
      "global_load_async_to_lds_b128 %0, %1, %2 offset:16"
      :: "v"(ldso), "v"(voff), "s"(sbase) : "memory");
}
// copy 64 contiguous bytes
DEVINL void async_cp64(unsigned ldso, unsigned voff, const void* sbase) {
  asm volatile(
      "global_load_async_to_lds_b128 %0, %1, %2\n\t"
      "global_load_async_to_lds_b128 %0, %1, %2 offset:16\n\t"
      "global_load_async_to_lds_b128 %0, %1, %2 offset:32\n\t"
      "global_load_async_to_lds_b128 %0, %1, %2 offset:48"
      :: "v"(ldso), "v"(voff), "s"(sbase) : "memory");
}
DEVINL void async_wait0() {
  asm volatile("s_wait_asynccnt 0x0" ::: "memory");
}

// ---------------------------------------------------------------- convert
__global__ void k_cvt(const float* __restrict__ s, __bf16* __restrict__ d, int n) {
  int i = blockIdx.x * blockDim.x + threadIdx.x;
  if (i < n) d[i] = f2bf(s[i]);
}

// ---------------------------------------------------------------- GEMM
// C[M,N] = A[M,K] * W[N,K]^T, bf16 inputs, f32 out. 128x128 tile, 8 waves.
constexpr int TM = 128, TN = 128, KSTEP = 32, LDT = 40;  // 8-elem pad vs banks

__global__ __launch_bounds__(256) void k_gemm(const __bf16* __restrict__ A,
                                              const __bf16* __restrict__ W,
                                              float* __restrict__ C,
                                              int M, int N, int K) {
  __shared__ __bf16 As[2][TM * LDT];
  __shared__ __bf16 Bs[2][TN * LDT];
  const int tid  = threadIdx.x;
  const int lane = tid & 31, wave = tid >> 5;
  const int ln   = lane & 15, hi = lane >> 4;
  const int wm   = wave >> 1, wn = wave & 1;           // 4x2 wave grid
  const int mt   = blockIdx.x * TM, nt0 = blockIdx.y * TN;
  const int frow = tid >> 1, fcol = (tid & 1) << 4;    // 16 bf16 per thread

  v8f acc[2][4];
  #pragma unroll
  for (int i = 0; i < 2; ++i)
    #pragma unroll
    for (int j = 0; j < 4; ++j) acc[i][j] = vzero();

  // per-thread global byte offsets (vaddr for GVS-mode async DMA)
  const unsigned gaoff = (unsigned)(((size_t)(mt + frow) * K + fcol) * 2);
  const unsigned gboff = (unsigned)(((size_t)(nt0 + frow) * K + fcol) * 2);
  const unsigned laoff0 = lds_off(&As[0][frow * LDT + fcol]);
  const unsigned lboff0 = lds_off(&Bs[0][frow * LDT + fcol]);
  const unsigned bufstride = (unsigned)(TM * LDT * 2);

  auto fill = [&](int buf, int k0) {   // async DMA: global -> LDS, no VGPRs
    const unsigned kb = (unsigned)(k0 * 2);
    async_cp32(laoff0 + (unsigned)buf * bufstride, gaoff + kb, A);
    async_cp32(lboff0 + (unsigned)buf * bufstride, gboff + kb, W);
  };

  auto compute = [&](int buf) {
    bf16x16 af[2], bw[4];
    #pragma unroll
    for (int mi = 0; mi < 2; ++mi)
      af[mi] = load_frag(&As[buf][(wm * 32 + mi * 16 + ln) * LDT], hi);
    #pragma unroll
    for (int ni = 0; ni < 4; ++ni)
      bw[ni] = load_frag(&Bs[buf][(wn * 64 + ni * 16 + ln) * LDT], hi);
    #pragma unroll
    for (int mi = 0; mi < 2; ++mi)
      #pragma unroll
      for (int ni = 0; ni < 4; ++ni)
        acc[mi][ni] = wmma_bf16(af[mi], bw[ni], acc[mi][ni]);
  };

  const int nk = K / KSTEP;
  fill(0, 0);
  async_wait0();
  __syncthreads();
  for (int kt = 0; kt < nk; ++kt) {
    const int cur = kt & 1;
    if (kt + 1 < nk) fill(cur ^ 1, (kt + 1) * KSTEP);   // DMA overlaps WMMAs
    if (kt + 2 < nk) {                                  // L2/WGP$ warm-up
      __builtin_prefetch((const char*)A + gaoff + (size_t)(kt + 2) * KSTEP * 2, 0, 3);
      __builtin_prefetch((const char*)W + gboff + (size_t)(kt + 2) * KSTEP * 2, 0, 3);
    }
    compute(cur);
    async_wait0();
    __syncthreads();
  }

  // C layout: VGPR j -> M=j (lanes 0-15) / M=j+8 (lanes 16-31), N=lane&15
  #pragma unroll
  for (int mi = 0; mi < 2; ++mi)
    #pragma unroll
    for (int ni = 0; ni < 4; ++ni)
      #pragma unroll
      for (int j = 0; j < 8; ++j) {
        int row = mt + wm * 32 + mi * 16 + j + hi * 8;
        int col = nt0 + wn * 64 + ni * 16 + ln;
        C[(size_t)row * N + col] = acc[mi][ni][j];
      }
}

// ---------------------------------------------------------- norm/rope/pack
// One wave per (token,head). mode 0: Q (rms+rope+gain+1/sqrt(HD)),
// mode 1: K (rms+rope), mode 2: V (transpose-pack only).
__global__ void k_normrope(const float* __restrict__ qkv,
                           const float* __restrict__ qgain,
                           __bf16* __restrict__ qbf,
                           __bf16* __restrict__ kbf,
                           __bf16* __restrict__ vtbf, int mode) {
  const int m = blockIdx.x;                 // token row
  const int h = threadIdx.x >> 5;
  const int lane = threadIdx.x & 31;
  const int b = m / SS, s = m % SS;
  const int col0 = (mode == 0 ? 0 : (mode == 1 ? DD : DD + KVHn * HDn)) + h * HDn;
  const float* src = qkv + (size_t)m * NQKV + col0;
  float t0 = src[lane], t1 = src[lane + 32], t2 = src[lane + 64], t3 = src[lane + 96];

  if (mode == 2) {                          // V^T pack: [b][kvh][d][s]
    __bf16* dst = vtbf + ((size_t)(b * KVHn + h) * HDn) * SS + s;
    dst[(size_t)lane * SS]        = f2bf(t0);
    dst[(size_t)(lane + 32) * SS] = f2bf(t1);
    dst[(size_t)(lane + 64) * SS] = f2bf(t2);
    dst[(size_t)(lane + 96) * SS] = f2bf(t3);
    return;
  }

  float ssum = t0 * t0 + t1 * t1 + t2 * t2 + t3 * t3;
  #pragma unroll
  for (int o = 16; o >= 1; o >>= 1) ssum += __shfl_xor(ssum, o, 32);
  const float r = rsqrtf(ssum * (1.0f / HDn) + 1.1920929e-07f);
  t0 *= r; t1 *= r; t2 *= r; t3 *= r;

  // RoPE: lane L holds the (d=L, d=L+32) pair; dims 64..127 pass through
  const float inv_freq = __expf(-(float)lane * (9.210340371976184f / 32.0f));
  const float fr = (float)s * inv_freq;
  const float cs = cosf(fr), sn = sinf(fr);
  const float n0 =  t0 * cs + t1 * sn;
  const float n1 = -t0 * sn + t1 * cs;
  t0 = n0; t1 = n1;

  float g = 1.0f;
  if (mode == 0) g = qgain[h] * 0.08838834764831845f;  // fold HD^-0.5 into Q
  t0 *= g; t1 *= g; t2 *= g; t3 *= g;

  __bf16* dst = (mode == 0)
      ? qbf + ((size_t)(b * HH + h) * SS + s) * HDn
      : kbf + ((size_t)(b * KVHn + h) * SS + s) * HDn;
  dst[lane] = f2bf(t0); dst[lane + 32] = f2bf(t1);
  dst[lane + 64] = f2bf(t2); dst[lane + 96] = f2bf(t3);
}

// ---------------------------------------------------------------- attention
// Dynamic LDS (MI455X WGP has 320KB): double-buffered K/V^T tiles so the
// async DMA for block kb+1 is hidden under block kb's WMMAs + softmax.
constexpr int QT = 128, KTL = 64;
constexpr int KROW = HDn + 8;                     // 136 bf16 per K-tile row
constexpr int VROW = KTL + 8;                     // 72 bf16 per V^T/P row
constexpr int KBUF = KTL * KROW;                  // elems per K buffer
constexpr int VBUF = HDn * VROW;                  // elems per V^T buffer
constexpr unsigned ATTN_LDS_BYTES =
    (2 * KBUF + 2 * VBUF + 8 * 16 * VROW) * 2;    // 90,112 B (3 WGs / WGP)

__global__ __launch_bounds__(256) void k_attn(const __bf16* __restrict__ qbf,
                                              const __bf16* __restrict__ kbf,
                                              const __bf16* __restrict__ vtbf,
                                              __bf16* __restrict__ ybf) {
  extern __shared__ __bf16 smem[];
  __bf16* const KsB = smem;                       // [2][KTL][KROW]
  __bf16* const VtB = smem + 2 * KBUF;            // [2][HDn][VROW]
  __bf16* const PsB = VtB + 2 * VBUF;             // [8][16][VROW]

  const int tid = threadIdx.x, lane = tid & 31, wave = tid >> 5;
  const int ln = lane & 15, hi = lane >> 4;
  const int bh = blockIdx.y, b = bh / HH, h = bh % HH, kvh = h / (HH / KVHn);
  const int q0 = blockIdx.x * QT;
  const int qrow = q0 + wave * 16 + ln;           // this lane's A-matrix row

  const __bf16* qptr = qbf + ((size_t)(b * HH + h) * SS + qrow) * HDn;
  bf16x16 qa[4];                                  // Q fragments resident in regs
  #pragma unroll
  for (int kt = 0; kt < 4; ++kt) qa[kt] = load_frag(qptr + kt * 32, hi);

  const __bf16* kbase = kbf + (size_t)(b * KVHn + kvh) * SS * HDn;
  const __bf16* vbase = vtbf + (size_t)(b * KVHn + kvh) * HDn * SS;

  // async-DMA staging addresses (per-thread, fixed across k-blocks)
  const int kr = tid >> 2, kc = (tid & 3) << 5;   // K tile: 64B per thread
  const int vr = tid >> 1, vc = (tid & 1) << 5;   // V^T tile: 64B per thread
  const unsigned kldso  = lds_off(&KsB[kr * KROW + kc]);
  const unsigned vldso  = lds_off(&VtB[vr * VROW + vc]);
  const unsigned kbufB  = (unsigned)(KBUF * 2);
  const unsigned vbufB  = (unsigned)(VBUF * 2);
  const unsigned kvoff0 = (unsigned)(((size_t)kr * HDn + kc) * 2);
  const unsigned vvoff0 = (unsigned)(((size_t)vr * SS + vc) * 2);

  auto stage = [&](int kb, int buf) {
    async_cp64(kldso + (unsigned)buf * kbufB, kvoff0 + (unsigned)(kb * KTL * HDn * 2), kbase);
    async_cp64(vldso + (unsigned)buf * vbufB, vvoff0 + (unsigned)(kb * KTL * 2),       vbase);
  };

  v8f o[8];
  #pragma unroll
  for (int nt = 0; nt < 8; ++nt) o[nt] = vzero();
  float mrow[8], lrow[8], alpha[8];
  #pragma unroll
  for (int j = 0; j < 8; ++j) { mrow[j] = -1e30f; lrow[j] = 0.f; }

  const int nkb = q0 / KTL + 2;                   // causal: keys <= q0+127
  stage(0, 0);
  async_wait0();
  __syncthreads();
  for (int kb = 0; kb < nkb; ++kb) {
    const int cur = kb & 1;
    if (kb + 1 < nkb) stage(kb + 1, cur ^ 1);     // DMA hides under compute
    const __bf16* Ks  = KsB + cur * KBUF;
    const __bf16* Vts = VtB + cur * VBUF;

    // S = Q * K^T  (16x64 per wave)
    v8f sacc[4];
    #pragma unroll
    for (int nt = 0; nt < 4; ++nt) {
      sacc[nt] = vzero();
      #pragma unroll
      for (int kt = 0; kt < 4; ++kt) {
        bf16x16 kf = load_frag(&Ks[(nt * 16 + ln) * KROW + kt * 32], hi);
        sacc[nt] = wmma_bf16(qa[kt], kf, sacc[nt]);
      }
    }

    // causal mask + online softmax (rows live in 16-lane halves)
    #pragma unroll
    for (int j = 0; j < 8; ++j) {
      const int mg = q0 + wave * 16 + j + hi * 8;
      float vmax = -1e30f;
      #pragma unroll
      for (int nt = 0; nt < 4; ++nt) {
        const int ng = kb * KTL + nt * 16 + ln;
        float sv = sacc[nt][j];
        if (ng > mg) sv = -1e30f;
        sacc[nt][j] = sv;
        vmax = fmaxf(vmax, sv);
      }
      #pragma unroll
      for (int o2 = 8; o2 >= 1; o2 >>= 1) vmax = fmaxf(vmax, __shfl_xor(vmax, o2, 32));
      const float mn = fmaxf(mrow[j], vmax);
      const float al = __expf(mrow[j] - mn);
      float rs = 0.f;
      #pragma unroll
      for (int nt = 0; nt < 4; ++nt) {
        const float p = __expf(sacc[nt][j] - mn);
        rs += p;
        PsB[(wave * 16 + j + hi * 8) * VROW + nt * 16 + ln] = f2bf(p);
      }
      #pragma unroll
      for (int o2 = 8; o2 >= 1; o2 >>= 1) rs += __shfl_xor(rs, o2, 32);
      lrow[j] = lrow[j] * al + rs;
      mrow[j] = mn;
      alpha[j] = al;
    }

    #pragma unroll
    for (int nt = 0; nt < 8; ++nt)
      #pragma unroll
      for (int j = 0; j < 8; ++j) o[nt][j] *= alpha[j];

    // O += P * V  (A from per-wave Ps, B from V^T tile)
    bf16x16 pa[2];
    #pragma unroll
    for (int kt = 0; kt < 2; ++kt)
      pa[kt] = load_frag(&PsB[(wave * 16 + ln) * VROW + kt * 32], hi);
    #pragma unroll
    for (int nt = 0; nt < 8; ++nt)
      #pragma unroll
      for (int kt = 0; kt < 2; ++kt) {
        bf16x16 vf = load_frag(&Vts[(nt * 16 + ln) * VROW + kt * 32], hi);
        o[nt] = wmma_bf16(pa[kt], vf, o[nt]);
      }

    async_wait0();                                // next buffers landed
    __syncthreads();                              // + all reads of cur done
  }

  // epilogue: normalize and write y in [token][h*128+d] bf16 for out-proj
  #pragma unroll
  for (int j = 0; j < 8; ++j) {
    const float inv = 1.0f / lrow[j];
    const int row = b * SS + q0 + wave * 16 + j + hi * 8;
    #pragma unroll
    for (int nt = 0; nt < 8; ++nt) {
      const int col = h * HDn + nt * 16 + ln;
      ybf[(size_t)row * DD + col] = f2bf(o[nt][j] * inv);
    }
  }
}

// ---------------------------------------------------------------- driver
extern "C" void kernel_launch(void* const* d_in, const int* in_sizes, int n_in,
                              void* d_out, int out_size, void* d_ws, size_t ws_size,
                              hipStream_t stream) {
  const float* x  = (const float*)d_in[0];
  const float* wq = (const float*)d_in[1];
  const float* wk = (const float*)d_in[2];
  const float* wv = (const float*)d_in[3];
  const float* wo = (const float*)d_in[4];
  const float* qg = (const float*)d_in[5];

  char* p = (char*)d_ws;
  __bf16* xbf  = (__bf16*)p; p += (size_t)BSr * DD * 2;
  __bf16* wqkv = (__bf16*)p; p += (size_t)NQKV * DD * 2;
  __bf16* wobf = (__bf16*)p; p += (size_t)DD * DD * 2;
  float*  qkvf = (float*)p;  p += (size_t)BSr * NQKV * 4;
  __bf16* qbf  = (__bf16*)p; p += (size_t)BB * HH * SS * HDn * 2;
  __bf16* kbf  = (__bf16*)p; p += (size_t)BB * KVHn * SS * HDn * 2;
  __bf16* vtbf = (__bf16*)p; p += (size_t)BB * KVHn * SS * HDn * 2;
  __bf16* ybf  = (__bf16*)p; p += (size_t)BSr * DD * 2;

  auto cvt = [&](const float* s, __bf16* d, int n) {
    k_cvt<<<(n + 255) / 256, 256, 0, stream>>>(s, d, n);
  };
  cvt(x,  xbf,  BSr * DD);
  cvt(wq, wqkv,                                   DD * DD);
  cvt(wk, wqkv + (size_t)DD * DD,                 KVHn * HDn * DD);
  cvt(wv, wqkv + (size_t)(DD + KVHn * HDn) * DD,  KVHn * HDn * DD);
  cvt(wo, wobf, DD * DD);

  // fused QKV projection: [8192,1536] = xbf[8192,1024] * Wqkv^T
  k_gemm<<<dim3(BSr / TM, NQKV / TN), 256, 0, stream>>>(xbf, wqkv, qkvf, BSr, NQKV, DD);

  k_normrope<<<BSr, 32 * HH,   0, stream>>>(qkvf, qg, qbf, kbf, vtbf, 0);  // Q
  k_normrope<<<BSr, 32 * KVHn, 0, stream>>>(qkvf, qg, qbf, kbf, vtbf, 1);  // K
  k_normrope<<<BSr, 32 * KVHn, 0, stream>>>(qkvf, qg, qbf, kbf, vtbf, 2);  // V^T

  k_attn<<<dim3(SS / QT, BB * HH), 256, ATTN_LDS_BYTES, stream>>>(qbf, kbf, vtbf, ybf);

  // out projection straight into fp32 d_out
  k_gemm<<<dim3(BSr / TM, DD / TN), 256, 0, stream>>>(ybf, wobf, (float*)d_out, BSr, DD, DD);
}
